// EnhancedBVHRouter_64390149702163
// MI455X (gfx1250) — matrix operand order repaired and checked by reading, not compiled, not run on hardware
//
#include <hip/hip_runtime.h>
#include <stdint.h>

// ---------------------------------------------------------------------------
// EnhancedBVHRouter for MI455X (gfx1250, wave32, WMMA bf16 16x16x32).
//
//   1. x (f32) -> xbf (bf16); pre-pack all weights into WMMA-B fragment order
//   2. h1   = GELU(xbf @ ipW1 + b1)     WMMA+async-LDS  (K=2048, N=512)
//   3. hpre = h1 @ ipW2 + b2            WMMA+async-LDS  (K=512,  N=256) -> f32
//   4. h    = LayerNorm(hpre)  -> bf16
//   5. 3 levels: feat GEMMs (WMMA direct) + scalar router (nc=4) -> probs
//   6. combined = [feat3|p1|p2|p3|pad] ; e1 = GELU(comb @ ehW1) WMMA+async-LDS
//   7. logits = e1 @ ehW2 (WMMA) ; softmax+argmax -> d_out
//
// Big GEMMs: block 256 thr = 8 waves, block tile 128M x 64N, each wave a
// 32x32 register tile (4 accumulators). A block-tile (128x32 bf16 = 8 KB)
// staged in LDS via GLOBAL_LOAD_ASYNC_TO_LDS_B128, double buffered,
// s_wait_asynccnt + barrier; copy of K-step k+1 overlaps WMMA of K-step k.
// ---------------------------------------------------------------------------

#define NB 32768      // batch rows
#define DIN 2048
#define FD 128

typedef __attribute__((ext_vector_type(16))) __bf16 bf16x16;
typedef __attribute__((ext_vector_type(8)))  float  f32x8;

__device__ __forceinline__ unsigned short f32_to_bf16(float f) {
  unsigned int u = __float_as_uint(f);
  u += 0x7FFFu + ((u >> 16) & 1u);          // round-to-nearest-even
  return (unsigned short)(u >> 16);
}
__device__ __forceinline__ float bf16_to_f32(unsigned short h) {
  return __uint_as_float(((unsigned int)h) << 16);
}
__device__ __forceinline__ float gelu_exact(float x) {
  return 0.5f * x * (1.0f + erff(x * 0.70710678118654752f));
}

// --- CDNA5 async copy: global -> LDS, 16B per lane, tracked by ASYNCcnt ----
__device__ __forceinline__ void async_load_b128(unsigned lds_addr, const void* gaddr) {
  asm volatile("global_load_async_to_lds_b128 %0, %1, off"
               :: "v"(lds_addr), "v"(gaddr) : "memory");
}
__device__ __forceinline__ void s_wait_asynccnt0() {
  asm volatile("s_wait_asynccnt 0x0" ::: "memory");
}

// ---------------------------------------------------------------------------
// f32 -> bf16 elementwise
__global__ void cvt_f32_bf16_kernel(const float* __restrict__ src,
                                    unsigned short* __restrict__ dst, int n) {
  int i = blockIdx.x * blockDim.x + threadIdx.x;
  if (i < n) dst[i] = f32_to_bf16(src[i]);
}

// ---------------------------------------------------------------------------
// Pre-pack W (K x N, f32 row-major) into WMMA B-fragment order, bf16.
// Tile (kt, nt) at ((nt*nkt + kt) << 9); lane l owns 16 contiguous bf16:
// element i = W[kt*32 + (i>>3)*16 + (l>>4)*8 + (i&7)][nt*16 + (l&15)]  (ISA 7.12.2).
__global__ void prepack_w_kernel(const float* __restrict__ W,
                                 unsigned short* __restrict__ Wp,
                                 int K, int N, int Kpad) {
  int idx = blockIdx.x * blockDim.x + threadIdx.x;
  int nkt = Kpad >> 5;
  int total = (N >> 4) * nkt * 512;
  if (idx >= total) return;
  int e    = idx & 15;
  int lane = (idx >> 4) & 31;
  int tile = idx >> 9;
  int kt   = tile % nkt;
  int nt   = tile / nkt;
  int krow = (kt << 5) + ((e >> 3) << 4) + ((lane >> 4) << 3) + (e & 7);
  int ncol = (nt << 4) + (lane & 15);
  float v = (krow < K) ? W[(size_t)krow * N + ncol] : 0.0f;
  Wp[idx] = f32_to_bf16(v);
}

// ---------------------------------------------------------------------------
// WMMA fragment loaders (wave32 layouts, CDNA5 ISA 7.12.2)

// A 16x32 bf16 tile from row-major activations.
__device__ __forceinline__ bf16x16 load_a_frag(const unsigned short* __restrict__ A,
                                               int ldA, int row, int k0, int lane) {
  const unsigned short* base = A + (size_t)row * ldA + k0 + ((lane >> 4) << 3);
  union { bf16x16 v; uint4 u[2]; } r;
  r.u[0] = *reinterpret_cast<const uint4*>(base);
  r.u[1] = *reinterpret_cast<const uint4*>(base + 16);
  return r.v;
}

// B 32x16 bf16 tile from pre-packed weights: 32 contiguous bytes per lane.
__device__ __forceinline__ bf16x16 load_b_frag(const unsigned short* __restrict__ Wp,
                                               int kt, int nt, int nkt, int lane) {
  const unsigned short* base = Wp + (((size_t)nt * nkt + kt) << 9) + (lane << 4);
  union { bf16x16 v; uint4 u[2]; } r;
  r.u[0] = *reinterpret_cast<const uint4*>(base);
  r.u[1] = *reinterpret_cast<const uint4*>(base + 8);
  return r.v;
}

// ---------------------------------------------------------------------------
// Big-GEMM kernel with async-LDS A staging.
// Block tile 128M x 64N; wave (mw=wave>>1, nw=wave&1) owns 32x32.
// Requires M%128==0, N%64==0, K%32==0. ldA=K, ldO=N.
template <int EPI>
__launch_bounds__(256)
__global__ void gemm_bf16_lds_kernel(const unsigned short* __restrict__ A,
                                     const unsigned short* __restrict__ Wp,
                                     const float* __restrict__ bias,
                                     int K,
                                     unsigned short* __restrict__ Obf,
                                     float* __restrict__ Of,
                                     int N) {
  // fragment-linear A tile: mtile t, lane l -> 32 contiguous bytes at
  // t*1024 + l*32 (segment s=0: elems 0..7, s=1: elems 8..15).
  __shared__ __align__(16) unsigned char shA[2][8192];

  const int tid  = threadIdx.x;
  const int lane = tid & 31;
  const int wave = tid >> 5;
  const int mw = wave >> 1, nw = wave & 1;
  const int rowblk = blockIdx.y * 128;
  const int ntb = blockIdx.x * 4 + nw * 2;
  const int nkt = K >> 5;

  // per-thread async-copy plan: 2 segments of 16B
  const unsigned short* gsrc[2];
  unsigned ldst[2];
  const unsigned shbase = (unsigned)(size_t)&shA[0][0];
#pragma unroll
  for (int j = 0; j < 2; ++j) {
    int seg = tid * 2 + j;              // 0..511
    int t = seg >> 6;                   // mtile 0..7
    int l = (seg >> 1) & 31;            // consumer lane
    int s = seg & 1;                    // segment
    int row = rowblk + t * 16 + (l & 15);
    int col = s * 16 + ((l >> 4) << 3); // K-offset inside the 32-wide tile
    gsrc[j] = A + (size_t)row * K + col;
    ldst[j] = (unsigned)(t * 1024 + l * 32 + s * 16);
  }

  f32x8 acc[2][2] = {};

  // prologue: stage kt=0 into buffer 0
#pragma unroll
  for (int j = 0; j < 2; ++j) async_load_b128(shbase + ldst[j], gsrc[j]);

  for (int kt = 0; kt < nkt; ++kt) {
    s_wait_asynccnt0();                 // my stage-kt copies done
    __syncthreads();                    // everyone's copies visible
    if (kt + 1 < nkt) {                 // overlap next copy with compute
      unsigned boff = ((kt + 1) & 1) ? 8192u : 0u;
#pragma unroll
      for (int j = 0; j < 2; ++j)
        async_load_b128(shbase + boff + ldst[j], gsrc[j] + ((kt + 1) << 5));
    }
    const int bufi = kt & 1;
    union { bf16x16 v; uint4 u[2]; } a0, a1;
    {
      const uint4* p0 = reinterpret_cast<const uint4*>(&shA[bufi][(mw * 2 + 0) * 1024 + lane * 32]);
      const uint4* p1 = reinterpret_cast<const uint4*>(&shA[bufi][(mw * 2 + 1) * 1024 + lane * 32]);
      a0.u[0] = p0[0]; a0.u[1] = p0[1];
      a1.u[0] = p1[0]; a1.u[1] = p1[1];
    }
    bf16x16 b0 = load_b_frag(Wp, kt, ntb + 0, nkt, lane);
    bf16x16 b1 = load_b_frag(Wp, kt, ntb + 1, nkt, lane);
    acc[0][0] = __builtin_amdgcn_wmma_f32_16x16x32_bf16(false, a0.v, false, b0, (short)0, acc[0][0], false, false);
    acc[0][1] = __builtin_amdgcn_wmma_f32_16x16x32_bf16(false, a0.v, false, b1, (short)0, acc[0][1], false, false);
    acc[1][0] = __builtin_amdgcn_wmma_f32_16x16x32_bf16(false, a1.v, false, b0, (short)0, acc[1][0], false, false);
    acc[1][1] = __builtin_amdgcn_wmma_f32_16x16x32_bf16(false, a1.v, false, b1, (short)0, acc[1][1], false, false);
    __syncthreads();                    // reads done before buffer reuse
  }

#pragma unroll
  for (int mi = 0; mi < 2; ++mi)
#pragma unroll
    for (int ni = 0; ni < 2; ++ni) {
      int col = ((ntb + ni) << 4) + (lane & 15);
      float bv = bias[col];
      int rbase = rowblk + (mw * 2 + mi) * 16 + ((lane >> 4) << 3);
#pragma unroll
      for (int r = 0; r < 8; ++r) {
        float v = acc[mi][ni][r] + bv;
        if (EPI == 1) v = gelu_exact(v);
        size_t o = (size_t)(rbase + r) * N + col;
        if (Obf) Obf[o] = f32_to_bf16(v);
        if (Of)  Of[o]  = v;
      }
    }
}

// ---------------------------------------------------------------------------
// Generic direct-load WMMA GEMM (small N). Block tile 64M x 32N, one 16x16
// tile per wave. ldA = K, ldO = N. EPI: 0 = identity, 1 = exact GELU.
template <int EPI>
__launch_bounds__(256)
__global__ void gemm_bf16_kernel(const unsigned short* __restrict__ A,
                                 const unsigned short* __restrict__ Wp,
                                 const float* __restrict__ bias,
                                 int K,
                                 unsigned short* __restrict__ Obf,
                                 float* __restrict__ Of,
                                 int N) {
  const int lane  = threadIdx.x & 31;
  const int wave  = threadIdx.x >> 5;
  const int mtile = blockIdx.y * 4 + (wave >> 1);
  const int ntile = blockIdx.x * 2 + (wave & 1);
  const int row0  = mtile << 4;
  const int nkt   = K >> 5;
  const int m     = lane & 15;

  f32x8 acc = {};
  for (int kt = 0; kt < nkt; ++kt) {
    bf16x16 a = load_a_frag(A, K, row0 + m, kt << 5, lane);
    bf16x16 b = load_b_frag(Wp, kt, ntile, nkt, lane);
    acc = __builtin_amdgcn_wmma_f32_16x16x32_bf16(false, a, false, b,
                                                  (short)0, acc, false, false);
  }

  const int col   = (ntile << 4) + (lane & 15);
  const float bv  = bias[col];
  const int rbase = row0 + ((lane >> 4) << 3);
#pragma unroll
  for (int r = 0; r < 8; ++r) {
    float v = acc[r] + bv;
    if (EPI == 1) v = gelu_exact(v);
    size_t o = (size_t)(rbase + r) * N + col;
    if (Obf) Obf[o] = f32_to_bf16(v);
    if (Of)  Of[o]  = v;
  }
}

// ---------------------------------------------------------------------------
// LayerNorm(256) over f32 rows -> bf16. One wave per row, 8 elems per lane.
__launch_bounds__(256)
__global__ void layernorm_kernel(const float* __restrict__ Hpre,
                                 const float* __restrict__ g,
                                 const float* __restrict__ bta,
                                 unsigned short* __restrict__ Hout) {
  const int wave = threadIdx.x >> 5, lane = threadIdx.x & 31;
  const int row  = blockIdx.x * 8 + wave;
  const float* p = Hpre + (size_t)row * 256;
  float vals[8], s = 0.f, ss = 0.f;
#pragma unroll
  for (int i = 0; i < 8; ++i) {
    float v = p[lane + i * 32];
    vals[i] = v; s += v; ss += v * v;
  }
  for (int off = 16; off; off >>= 1) {
    s  += __shfl_xor(s,  off, 32);
    ss += __shfl_xor(ss, off, 32);
  }
  const float mu  = s * (1.0f / 256.0f);
  const float var = ss * (1.0f / 256.0f) - mu * mu;
  const float inv = rsqrtf(var + 1e-5f);
  unsigned short* o = Hout + (size_t)row * 256;
#pragma unroll
  for (int i = 0; i < 8; ++i) {
    int c = lane + i * 32;
    o[c] = f32_to_bf16((vals[i] - mu) * inv * g[c] + bta[c]);
  }
}

// ---------------------------------------------------------------------------
// Router head for one level (nc = 4): one thread per row.
__launch_bounds__(256)
__global__ void router_kernel(const unsigned short* __restrict__ in, int din,
                              const unsigned short* __restrict__ feat,
                              const float* __restrict__ W3d,
                              const float* __restrict__ b3d,
                              const float* __restrict__ cen,
                              const float* __restrict__ rW,   // (FD+3) x 4
                              const float* __restrict__ rb,
                              float* __restrict__ probs) {
  const int row = blockIdx.x * blockDim.x + threadIdx.x;
  const unsigned short* xr = in + (size_t)row * din;
  float pos[3] = {b3d[0], b3d[1], b3d[2]};
  for (int k = 0; k < din; ++k) {
    float xv = bf16_to_f32(xr[k]);
    pos[0] += xv * W3d[k * 3 + 0];
    pos[1] += xv * W3d[k * 3 + 1];
    pos[2] += xv * W3d[k * 3 + 2];
  }
  const unsigned short* fr = feat + (size_t)row * FD;
  float logit[4];
#pragma unroll
  for (int c = 0; c < 4; ++c) {
    float dsq = 0.f;
#pragma unroll
    for (int d = 0; d < 3; ++d) { float t = pos[d] - cen[c * 3 + d]; dsq += t * t; }
    float acc = rb[c] - 0.5f * dsq * (1.0f / (2.0f + 1e-8f));
    for (int k = 0; k < FD; ++k) acc += bf16_to_f32(fr[k]) * rW[k * 4 + c];
    acc += pos[0] * rW[(FD + 0) * 4 + c]
         + pos[1] * rW[(FD + 1) * 4 + c]
         + pos[2] * rW[(FD + 2) * 4 + c];
    logit[c] = acc;
  }
  float mx = fmaxf(fmaxf(logit[0], logit[1]), fmaxf(logit[2], logit[3]));
  float e[4], se = 0.f;
#pragma unroll
  for (int c = 0; c < 4; ++c) { e[c] = __expf(logit[c] - mx); se += e[c]; }
  float is = 1.0f / se;
#pragma unroll
  for (int c = 0; c < 4; ++c) probs[(size_t)row * 4 + c] = e[c] * is;
}

// ---------------------------------------------------------------------------
// combined (B x 160 bf16) = [feat3(128) | p1(4) | p2(4) | p3(4) | zeros(20)]
__global__ void combined_kernel(const unsigned short* __restrict__ f3,
                                const float* __restrict__ p1,
                                const float* __restrict__ p2,
                                const float* __restrict__ p3,
                                unsigned short* __restrict__ out) {
  int idx = blockIdx.x * blockDim.x + threadIdx.x;
  if (idx >= NB * 160) return;
  int row = idx / 160, c = idx - row * 160;
  unsigned short v;
  if (c < 128)      v = f3[(size_t)row * FD + c];
  else if (c < 132) v = f32_to_bf16(p1[(size_t)row * 4 + (c - 128)]);
  else if (c < 136) v = f32_to_bf16(p2[(size_t)row * 4 + (c - 132)]);
  else if (c < 140) v = f32_to_bf16(p3[(size_t)row * 4 + (c - 136)]);
  else              v = 0;
  out[idx] = v;
}

// ---------------------------------------------------------------------------
// Final softmax(64) + argmax: one wave per row, 2 cols per lane.
__launch_bounds__(256)
__global__ void softmax_argmax_kernel(const float* __restrict__ logits,
                                      float* __restrict__ probs,
                                      int* __restrict__ ids) {
  const int wave = threadIdx.x >> 5, lane = threadIdx.x & 31;
  const int row  = blockIdx.x * 8 + wave;
  const float* lr = logits + (size_t)row * 64;
  float v0 = lr[lane], v1 = lr[lane + 32];
  float mx = fmaxf(v0, v1);
  for (int off = 16; off; off >>= 1) mx = fmaxf(mx, __shfl_xor(mx, off, 32));
  float e0 = __expf(v0 - mx), e1 = __expf(v1 - mx);
  float s = e0 + e1;
  for (int off = 16; off; off >>= 1) s += __shfl_xor(s, off, 32);
  float is = 1.0f / s;
  probs[(size_t)row * 64 + lane]      = e0 * is;
  probs[(size_t)row * 64 + lane + 32] = e1 * is;
  float bv = v0; int bi = lane;
  if (v1 > bv) { bv = v1; bi = lane + 32; }
  for (int off = 16; off; off >>= 1) {
    float ov = __shfl_xor(bv, off, 32);
    int   oi = __shfl_xor(bi, off, 32);
    if (ov > bv || (ov == bv && oi < bi)) { bv = ov; bi = oi; }
  }
  if (lane == 0) ids[row] = bi;
}

// ---------------------------------------------------------------------------
extern "C" void kernel_launch(void* const* d_in, const int* in_sizes, int n_in,
                              void* d_out, int out_size, void* d_ws, size_t ws_size,
                              hipStream_t stream) {
  const float* x    = (const float*)d_in[0];
  const float* ipW1 = (const float*)d_in[1];
  const float* ipb1 = (const float*)d_in[2];
  const float* ipW2 = (const float*)d_in[3];
  const float* ipb2 = (const float*)d_in[4];
  const float* ln_g = (const float*)d_in[5];
  const float* ln_b = (const float*)d_in[6];
  const float* L[3][10];  // W3d, b3d, c, lr, f1W, f1b, f2W, f2b, rW, rb
  for (int l = 0; l < 3; ++l)
    for (int j = 0; j < 10; ++j)
      L[l][j] = (const float*)d_in[7 + l * 10 + j];
  const float* ehW1 = (const float*)d_in[37];
  const float* ehb1 = (const float*)d_in[38];
  const float* ehW2 = (const float*)d_in[39];
  const float* ehb2 = (const float*)d_in[40];

  char* base = (char*)d_ws;
  size_t off = 0;
  auto alloc = [&](size_t bytes) -> void* {
    void* p = base + off;
    off = (off + bytes + 255) & ~(size_t)255;
    return p;
  };
  unsigned short* xbf   = (unsigned short*)alloc((size_t)NB * DIN * 2);
  unsigned short* h1    = (unsigned short*)alloc((size_t)NB * 512 * 2);
  float*          hpre  = (float*)         alloc((size_t)NB * 256 * 4);
  unsigned short* h     = (unsigned short*)alloc((size_t)NB * 256 * 2);
  unsigned short* tbuf  = (unsigned short*)alloc((size_t)NB * FD * 2);
  unsigned short* feat1 = (unsigned short*)alloc((size_t)NB * FD * 2);
  unsigned short* feat2 = (unsigned short*)alloc((size_t)NB * FD * 2);
  unsigned short* feat3 = (unsigned short*)alloc((size_t)NB * FD * 2);
  float*          pr1   = (float*)         alloc((size_t)NB * 4 * 4);
  float*          pr2   = (float*)         alloc((size_t)NB * 4 * 4);
  float*          pr3   = (float*)         alloc((size_t)NB * 4 * 4);
  unsigned short* comb  = (unsigned short*)alloc((size_t)NB * 160 * 2);
  unsigned short* e1    = (unsigned short*)alloc((size_t)NB * 256 * 2);
  float*          lg2   = (float*)         alloc((size_t)NB * 64 * 4);
  unsigned short* ipW1p = (unsigned short*)alloc((size_t)2048 * 512 * 2);
  unsigned short* ipW2p = (unsigned short*)alloc((size_t)512 * 256 * 2);
  unsigned short* f1Wp[3], *f2Wp[3];
  f1Wp[0] = (unsigned short*)alloc((size_t)256 * FD * 2);
  f2Wp[0] = (unsigned short*)alloc((size_t)FD * FD * 2);
  f1Wp[1] = (unsigned short*)alloc((size_t)FD * FD * 2);
  f2Wp[1] = (unsigned short*)alloc((size_t)FD * FD * 2);
  f1Wp[2] = (unsigned short*)alloc((size_t)FD * FD * 2);
  f2Wp[2] = (unsigned short*)alloc((size_t)FD * FD * 2);
  unsigned short* ehW1p = (unsigned short*)alloc((size_t)160 * 256 * 2);
  unsigned short* ehW2p = (unsigned short*)alloc((size_t)256 * 64 * 2);

  auto prepack = [&](const float* W, unsigned short* Wp, int K, int N, int Kpad) {
    int total = (N / 16) * (Kpad / 32) * 512;
    prepack_w_kernel<<<(total + 255) / 256, 256, 0, stream>>>(W, Wp, K, N, Kpad);
  };

  {
    int n = NB * DIN;
    cvt_f32_bf16_kernel<<<(n + 255) / 256, 256, 0, stream>>>(x, xbf, n);
  }
  prepack(ipW1, ipW1p, 2048, 512, 2048);
  prepack(ipW2, ipW2p, 512, 256, 512);
  prepack(L[0][4], f1Wp[0], 256, FD, 256);
  prepack(L[0][6], f2Wp[0], FD, FD, FD);
  prepack(L[1][4], f1Wp[1], FD, FD, FD);
  prepack(L[1][6], f2Wp[1], FD, FD, FD);
  prepack(L[2][4], f1Wp[2], FD, FD, FD);
  prepack(L[2][6], f2Wp[2], FD, FD, FD);
  prepack(ehW1, ehW1p, 140, 256, 160);   // pad K 140 -> 160
  prepack(ehW2, ehW2p, 256, 64, 256);

#define GEMM_LDS(EPI, A, Wp, bias, N, K, Obf, Of)                               \
  gemm_bf16_lds_kernel<EPI><<<dim3((N) / 64, NB / 128), dim3(256), 0, stream>>>(\
      (A), (Wp), (bias), (K), (Obf), (Of), (N))
#define GEMM(EPI, A, Wp, bias, N, K, Obf, Of)                                   \
  gemm_bf16_kernel<EPI><<<dim3((N) / 32, NB / 64), dim3(256), 0, stream>>>(     \
      (A), (Wp), (bias), (K), (Obf), (Of), (N))

  // input_proj
  GEMM_LDS(1, xbf, ipW1p, ipb1, 512, 2048, h1, (float*)nullptr);
  GEMM_LDS(0, h1, ipW2p, ipb2, 256, 512, (unsigned short*)nullptr, hpre);
  layernorm_kernel<<<NB / 8, 256, 0, stream>>>(hpre, ln_g, ln_b, h);

  // levels
  const unsigned short* lin[3]  = {h, feat1, feat2};
  unsigned short*       lout[3] = {feat1, feat2, feat3};
  float*                lpr[3]  = {pr1, pr2, pr3};
  const int ldin[3] = {256, FD, FD};
  for (int l = 0; l < 3; ++l) {
    GEMM(1, lin[l], f1Wp[l], L[l][5], FD, ldin[l], tbuf, (float*)nullptr);
    GEMM(0, tbuf, f2Wp[l], L[l][7], FD, FD, lout[l], (float*)nullptr);
    router_kernel<<<NB / 256, 256, 0, stream>>>(lin[l], ldin[l], lout[l],
                                                L[l][0], L[l][1], L[l][2],
                                                L[l][8], L[l][9], lpr[l]);
  }

  // expert head
  {
    int n = NB * 160;
    combined_kernel<<<(n + 255) / 256, 256, 0, stream>>>(feat3, pr1, pr2, pr3, comb);
  }
  GEMM_LDS(1, comb, ehW1p, ehb1, 256, 160, e1, (float*)nullptr);
  GEMM(0, e1, ehW2p, ehb2, 64, 256, (unsigned short*)nullptr, lg2);

  float* probs_out = (float*)d_out;
  int*   ids_out   = (int*)((float*)d_out + (size_t)NB * 64);
  softmax_argmax_kernel<<<NB / 8, 256, 0, stream>>>(lg2, probs_out, ids_out);

#undef GEMM
#undef GEMM_LDS
  (void)in_sizes; (void)n_in; (void)out_size; (void)ws_size;
}
